// MoE_91293824844254
// MI455X (gfx1250) — matrix operand order repaired and checked by reading, not compiled, not run on hardware
//
#include <hip/hip_runtime.h>

// MoE forward for gfx1250 (MI455X): bf16 WMMA GEMMs, fused SwiGLU, top-2 router.
// T=4096 tokens, H=1024, I=4096, E=8. Dense-over-experts (matches reference math:
// w_full is 0 for unselected experts). Workspace: ~34 MB
//   [0, 128KB)      w_full   float[T][E]
//   [128KB, +64B)   expert_load[8], tok_cnt[8]
//   [131584, +32MB) hdn      bf16[T][I]   (reused per expert; fits in 192MB L2)
//
// v3: packed v_cvt_pk_bf16_f32 via native 2-wide __bf16 vectors (no repack movs),
//     B tiles staged as K-pairs -> one ds_store_b32 per (n, k-pair) instead of
//     two ds_store_b16; ping-pong LDS; async global->LDS DMA for GEMM2's A tile.

#define T_TOK 4096
#define H_DIM 1024
#define I_DIM 4096
#define E_NUM 8
#define K_TOP 2
#define AUX_COEF 0.001f

#define BM 64      // block tile M (tokens)
#define BN 128     // block tile N
#define BK 32      // K per WMMA step (bf16 16x16x32)
#define BSTR 48    // padded K-stride for N-major B tiles in LDS (16B-aligned rows)

#define HDN_OFF 131584  // byte offset of hdn in workspace (16B aligned)

typedef __bf16 bf16_t;
typedef bf16_t v2bf  __attribute__((ext_vector_type(2)));
typedef bf16_t v16bf __attribute__((ext_vector_type(16)));
typedef float  v8f   __attribute__((ext_vector_type(8)));
typedef float  f32x4 __attribute__((ext_vector_type(4)));
typedef unsigned int u32x4 __attribute__((ext_vector_type(4)));

union FragBF {
    v16bf v;
    u32x4 q[2];
};

// fp32 pair -> packed bf16 dword (single v_cvt_pk_bf16_f32)
__device__ __forceinline__ unsigned int bfpack2(float a, float b) {
    union { v2bf v; unsigned int u; } cv;
    v2bf t;
    t.x = (__bf16)a;
    t.y = (__bf16)b;
    cv.v = t;
    return cv.u;
}
__device__ __forceinline__ unsigned short bf1(float f) {
    union { __bf16 h; unsigned short u; } cv;
    cv.h = (__bf16)f;
    return cv.u;
}

__device__ __forceinline__ v8f wmma_bf16(v16bf a, v16bf b, v8f c) {
    // D = A(16x32 bf16) x B(32x16 bf16) + C(16x16 f32)
    return __builtin_amdgcn_wmma_f32_16x16x32_bf16(
        false, a, false, b, (short)0, c, false, false);
}

// async global -> LDS copy, 16B per active lane (ASYNCcnt-tracked)
__device__ __forceinline__ void async_copy_b128(unsigned lds_off, const void* gptr) {
    asm volatile("global_load_async_to_lds_b128 %0, %1, off"
                 :: "v"(lds_off), "v"(gptr) : "memory");
}
__device__ __forceinline__ void wait_async0() {
    asm volatile("s_wait_asynccnt 0x0" ::: "memory");
}
__device__ __forceinline__ unsigned lds_addr_of(const void* p) {
    // generic LDS pointer: low 32 bits are the LDS byte address
    return (unsigned)(unsigned long long)p;
}

// ---------------------------------------------------------------- init
__global__ void moe_init_kernel(float* __restrict__ out, float* __restrict__ acc16) {
    const size_t n = (size_t)T_TOK * H_DIM + 1;
    for (size_t i = (size_t)blockIdx.x * blockDim.x + threadIdx.x; i < n;
         i += (size_t)gridDim.x * blockDim.x)
        out[i] = 0.0f;
    if (blockIdx.x == 0 && threadIdx.x < 16) acc16[threadIdx.x] = 0.0f;
}

// ---------------------------------------------------------------- router
// one wave32 per token; 8 waves per block
__global__ __launch_bounds__(256) void moe_router_kernel(
    const float* __restrict__ x, const float* __restrict__ gw,
    float* __restrict__ w_full, float* __restrict__ expert_load,
    float* __restrict__ tok_cnt)
{
    const int wave = threadIdx.x >> 5;
    const int lane = threadIdx.x & 31;
    const int t = blockIdx.x * 8 + wave;
    if (t >= T_TOK) return;

    float acc[E_NUM];
#pragma unroll
    for (int e = 0; e < E_NUM; ++e) acc[e] = 0.0f;

    const float* xr = x + (size_t)t * H_DIM;
    for (int k = lane; k < H_DIM; k += 32) {
        const float xv = xr[k];
        const float* g = gw + (size_t)k * E_NUM;
#pragma unroll
        for (int e = 0; e < E_NUM; ++e) acc[e] = fmaf(xv, g[e], acc[e]);
    }
#pragma unroll
    for (int off = 16; off >= 1; off >>= 1) {
#pragma unroll
        for (int e = 0; e < E_NUM; ++e) acc[e] += __shfl_xor(acc[e], off, 32);
    }

    if (lane == 0) {
        float m = acc[0];
#pragma unroll
        for (int e = 1; e < E_NUM; ++e) m = fmaxf(m, acc[e]);
        float rw[E_NUM];
        float s = 0.0f;
#pragma unroll
        for (int e = 0; e < E_NUM; ++e) { rw[e] = __expf(acc[e] - m); s += rw[e]; }
        const float inv = 1.0f / s;
#pragma unroll
        for (int e = 0; e < E_NUM; ++e) rw[e] *= inv;

        // top-2 (deterministic scan order)
        int i1 = 0;
#pragma unroll
        for (int e = 1; e < E_NUM; ++e) if (rw[e] > rw[i1]) i1 = e;
        int i2 = (i1 == 0) ? 1 : 0;
#pragma unroll
        for (int e = 0; e < E_NUM; ++e) if (e != i1 && rw[e] > rw[i2]) i2 = e;

        const float wsum = rw[i1] + rw[i2];
        const float w1 = rw[i1] / wsum;
        const float w2 = rw[i2] / wsum;
#pragma unroll
        for (int e = 0; e < E_NUM; ++e)
            w_full[(size_t)t * E_NUM + e] = (e == i1) ? w1 : ((e == i2) ? w2 : 0.0f);

        atomicAdd(&expert_load[i1], rw[i1]);
        atomicAdd(&expert_load[i2], rw[i2]);
        atomicAdd(&tok_cnt[i1], 1.0f);
        atomicAdd(&tok_cnt[i2], 1.0f);
    }
}

// ---------------------------------------------------------------- aux loss
__global__ void moe_aux_kernel(const float* __restrict__ expert_load,
                               const float* __restrict__ tok_cnt,
                               float* __restrict__ out_aux)
{
    if (threadIdx.x == 0 && blockIdx.x == 0) {
        float s = 0.0f;
        for (int e = 0; e < E_NUM; ++e)
            s += (expert_load[e] / (float)T_TOK) *
                 (tok_cnt[e] / (float)(T_TOK * K_TOP));
        *out_aux = s * (float)E_NUM * AUX_COEF;
    }
}

// ================================================================ GEMM1
// hdn[T,I] = silu(x @ wg[e]) * (x @ wu[e]), stored as bf16.
// 64x128 block tile, 8 waves x (32x32) wave tile, 2x2 WMMA subtiles,
// gate & up share the A tile. Ping-pong double-buffered LDS.
// B staging: each thread owns K rows {krow, krow+1} x 8 N columns, so each
// (n, k-pair) is one v_cvt_pk_bf16_f32 + one ds_store_b32.
struct G1Regs { f32x4 a[2]; f32x4 g[4]; f32x4 u[4]; };

__device__ __forceinline__ void g1_load(G1Regs& r, const float* __restrict__ x,
                                        const float* __restrict__ wgp,
                                        const float* __restrict__ wup,
                                        int m0, int n0, int kk,
                                        int arow, int ac0, int krow, int nc0) {
    const float* as = x + (size_t)(m0 + arow) * H_DIM + kk + ac0;
    r.a[0] = *(const f32x4*)as;
    r.a[1] = *(const f32x4*)(as + 4);
    const float* g0 = wgp + (size_t)(kk + krow) * I_DIM + n0 + nc0;
    const float* u0 = wup + (size_t)(kk + krow) * I_DIM + n0 + nc0;
    r.g[0] = *(const f32x4*)g0;
    r.g[1] = *(const f32x4*)(g0 + 4);
    r.g[2] = *(const f32x4*)(g0 + I_DIM);
    r.g[3] = *(const f32x4*)(g0 + I_DIM + 4);
    r.u[0] = *(const f32x4*)u0;
    r.u[1] = *(const f32x4*)(u0 + 4);
    r.u[2] = *(const f32x4*)(u0 + I_DIM);
    r.u[3] = *(const f32x4*)(u0 + I_DIM + 4);
}

__device__ __forceinline__ void g1_store(const G1Regs& r,
                                         unsigned short (*As)[BK],
                                         unsigned short (*Bg)[BSTR],
                                         unsigned short (*Bu)[BSTR],
                                         int arow, int ac0, int krow, int nc0) {
    u32x4 p;
    p.x = bfpack2(r.a[0].x, r.a[0].y);
    p.y = bfpack2(r.a[0].z, r.a[0].w);
    p.z = bfpack2(r.a[1].x, r.a[1].y);
    p.w = bfpack2(r.a[1].z, r.a[1].w);
    *(u32x4*)&As[arow][ac0] = p;
    const float* g0 = (const float*)&r.g[0];   // row k,   8 floats
    const float* g1 = (const float*)&r.g[2];   // row k+1, 8 floats
    const float* u0 = (const float*)&r.u[0];
    const float* u1 = (const float*)&r.u[2];
#pragma unroll
    for (int j = 0; j < 8; ++j) {
        *(unsigned int*)&Bg[nc0 + j][krow] = bfpack2(g0[j], g1[j]);
        *(unsigned int*)&Bu[nc0 + j][krow] = bfpack2(u0[j], u1[j]);
    }
}

__global__ __launch_bounds__(256) void moe_gemm1_kernel(
    const float* __restrict__ x, const float* __restrict__ wg,
    const float* __restrict__ wu, unsigned short* __restrict__ hdn, int e)
{
    __shared__ __align__(16) unsigned short As[2][BM][BK];
    __shared__ __align__(16) unsigned short Bg[2][BN][BSTR];
    __shared__ __align__(16) unsigned short Bu[2][BN][BSTR];

    const float* __restrict__ wgp = wg + (size_t)e * H_DIM * I_DIM;
    const float* __restrict__ wup = wu + (size_t)e * H_DIM * I_DIM;

    const int n0 = blockIdx.x * BN;
    const int m0 = blockIdx.y * BM;
    const int tid = threadIdx.x;
    const int wave = tid >> 5;
    const int lane = tid & 31;
    const int wm = (wave & 1) * 32;
    const int wn = (wave >> 1) * 32;
    const int lhalf = lane >> 4;      // 0: lanes 0-15, 1: lanes 16-31
    const int lrow = lane & 15;

    const int arow = tid >> 2;        // 0..63
    const int ac0 = (tid & 3) * 8;    // 0,8,16,24
    const int krow = (tid >> 4) * 2;  // 0,2,..,30 : K-pair owned by this thread
    const int nc0 = (tid & 15) * 8;   // 0..120    : 8 N columns owned

    v8f cg[2][2] = {};
    v8f cu[2][2] = {};

    // prologue: stage K tile 0 into buffer 0
    {
        G1Regs r;
        g1_load(r, x, wgp, wup, m0, n0, 0, arow, ac0, krow, nc0);
        g1_store(r, As[0], Bg[0], Bu[0], arow, ac0, krow, nc0);
    }
    __syncthreads();

    for (int kk = 0; kk < H_DIM; kk += BK) {
        const int cur = (kk >> 5) & 1;
        const int nxt = cur ^ 1;
        const bool has_next = (kk + BK) < H_DIM;

        // issue next-tile global loads into registers (overlap with WMMA)
        G1Regs r;
        if (has_next) {
            g1_load(r, x, wgp, wup, m0, n0, kk + BK, arow, ac0, krow, nc0);
            __builtin_prefetch(wgp + (size_t)(kk + 2 * BK + krow) * I_DIM + n0 + nc0, 0, 1);
            __builtin_prefetch(wup + (size_t)(kk + 2 * BK + krow) * I_DIM + n0 + nc0, 0, 1);
        }

        // fragments from current buffer (per ISA 16-bit A/B VGPR layouts)
        FragBF a[2], bg[2], bu[2];
#pragma unroll
        for (int sm = 0; sm < 2; ++sm) {
            const int rr = wm + sm * 16 + lrow;
            const int k0 = lhalf * 8;   // lanes 0-15: K{0..7,16..23}; 16-31: K{8..15,24..31}
            a[sm].q[0] = *(const u32x4*)&As[cur][rr][k0];
            a[sm].q[1] = *(const u32x4*)&As[cur][rr][k0 + 16];
        }
#pragma unroll
        for (int sn = 0; sn < 2; ++sn) {
            const int n = wn + sn * 16 + lrow;
            const int k0 = lhalf * 8;   // lanes 0-15: K{0..7,16..23}; 16-31: K{8..15,24..31}
            bg[sn].q[0] = *(const u32x4*)&Bg[cur][n][k0];
            bg[sn].q[1] = *(const u32x4*)&Bg[cur][n][k0 + 16];
            bu[sn].q[0] = *(const u32x4*)&Bu[cur][n][k0];
            bu[sn].q[1] = *(const u32x4*)&Bu[cur][n][k0 + 16];
        }
#pragma unroll
        for (int sm = 0; sm < 2; ++sm)
#pragma unroll
            for (int sn = 0; sn < 2; ++sn) {
                cg[sm][sn] = wmma_bf16(a[sm].v, bg[sn].v, cg[sm][sn]);
                cu[sm][sn] = wmma_bf16(a[sm].v, bu[sn].v, cu[sm][sn]);
            }

        // stage next tile into the other buffer
        if (has_next)
            g1_store(r, As[nxt], Bg[nxt], Bu[nxt], arow, ac0, krow, nc0);
        __syncthreads();
    }

    // epilogue: hdn = silu(gate) * up, store bf16
#pragma unroll
    for (int sm = 0; sm < 2; ++sm)
#pragma unroll
        for (int sn = 0; sn < 2; ++sn)
#pragma unroll
            for (int r = 0; r < 8; ++r) {
                const int row = m0 + wm + sm * 16 + r + (lhalf ? 8 : 0);
                const int col = n0 + wn + sn * 16 + lrow;
                const float g = cg[sm][sn][r];
                const float u = cu[sm][sn][r];
                const float sg_ = g / (1.0f + __expf(-g));   // silu
                hdn[(size_t)row * I_DIM + col] = bf1(sg_ * u);
            }
}

// ================================================================ GEMM2
// out[T,H] += (hdn @ wd[e]) * w_full[:, e]
// A tile (bf16 hdn) staged via async global->LDS DMA; B via regs (fp32->bf16,
// K-pair staging -> packed cvt + ds_store_b32).
struct G2Regs { f32x4 b[4]; };

__global__ __launch_bounds__(256) void moe_gemm2_kernel(
    const unsigned short* __restrict__ hdn, const float* __restrict__ wd,
    const float* __restrict__ w_full, float* __restrict__ out, int e)
{
    __shared__ __align__(16) unsigned short As[2][BM][BK];
    __shared__ __align__(16) unsigned short Bs[2][BN][BSTR];

    const float* __restrict__ wdp = wd + (size_t)e * I_DIM * H_DIM;

    const int n0 = blockIdx.x * BN;
    const int m0 = blockIdx.y * BM;
    const int tid = threadIdx.x;
    const int wave = tid >> 5;
    const int lane = tid & 31;
    const int wm = (wave & 1) * 32;
    const int wn = (wave >> 1) * 32;
    const int lhalf = lane >> 4;
    const int lrow = lane & 15;

    const int arow = tid >> 2;
    const int ac0 = (tid & 3) * 8;
    const int krow = (tid >> 4) * 2;
    const int nc0 = (tid & 15) * 8;

    v8f c[2][2] = {};

    // prologue: async A tile 0, B tile 0 via regs
    async_copy_b128(lds_addr_of(&As[0][arow][ac0]),
                    hdn + (size_t)(m0 + arow) * I_DIM + ac0);
    {
        const float* s = wdp + (size_t)krow * H_DIM + n0 + nc0;
        const float* s1 = s + H_DIM;
#pragma unroll
        for (int j = 0; j < 8; ++j)
            *(unsigned int*)&Bs[0][nc0 + j][krow] = bfpack2(s[j], s1[j]);
    }
    wait_async0();
    __syncthreads();

    for (int kk = 0; kk < I_DIM; kk += BK) {
        const int cur = (kk >> 5) & 1;
        const int nxt = cur ^ 1;
        const bool has_next = (kk + BK) < I_DIM;

        G2Regs r;
        if (has_next) {
            // async DMA of next bf16 A tile straight into LDS
            async_copy_b128(lds_addr_of(&As[nxt][arow][ac0]),
                            hdn + (size_t)(m0 + arow) * I_DIM + kk + BK + ac0);
            const float* s = wdp + (size_t)(kk + BK + krow) * H_DIM + n0 + nc0;
            r.b[0] = *(const f32x4*)s;
            r.b[1] = *(const f32x4*)(s + 4);
            r.b[2] = *(const f32x4*)(s + H_DIM);
            r.b[3] = *(const f32x4*)(s + H_DIM + 4);
            __builtin_prefetch(wdp + (size_t)(kk + 2 * BK + krow) * H_DIM + n0 + nc0, 0, 1);
        }

        FragBF a[2], b[2];
#pragma unroll
        for (int sm = 0; sm < 2; ++sm) {
            const int rr = wm + sm * 16 + lrow;
            const int k0 = lhalf * 8;
            a[sm].q[0] = *(const u32x4*)&As[cur][rr][k0];
            a[sm].q[1] = *(const u32x4*)&As[cur][rr][k0 + 16];
        }
#pragma unroll
        for (int sn = 0; sn < 2; ++sn) {
            const int n = wn + sn * 16 + lrow;
            const int k0 = lhalf * 8;
            b[sn].q[0] = *(const u32x4*)&Bs[cur][n][k0];
            b[sn].q[1] = *(const u32x4*)&Bs[cur][n][k0 + 16];
        }
#pragma unroll
        for (int sm = 0; sm < 2; ++sm)
#pragma unroll
            for (int sn = 0; sn < 2; ++sn)
                c[sm][sn] = wmma_bf16(a[sm].v, b[sn].v, c[sm][sn]);

        if (has_next) {
            const float* b0 = (const float*)&r.b[0];   // row k
            const float* b1 = (const float*)&r.b[2];   // row k+1
#pragma unroll
            for (int j = 0; j < 8; ++j)
                *(unsigned int*)&Bs[nxt][nc0 + j][krow] = bfpack2(b0[j], b1[j]);
        }
        wait_async0();     // next A tile resident before the barrier releases
        __syncthreads();
    }

    // epilogue: scale by per-token combine weight, accumulate into out
#pragma unroll
    for (int sm = 0; sm < 2; ++sm)
#pragma unroll
        for (int sn = 0; sn < 2; ++sn)
#pragma unroll
            for (int r = 0; r < 8; ++r) {
                const int row = m0 + wm + sm * 16 + r + (lhalf ? 8 : 0);
                const int col = n0 + wn + sn * 16 + lrow;
                const float scale = w_full[(size_t)row * E_NUM + e];
                out[(size_t)row * H_DIM + col] += scale * c[sm][sn][r];
            }
}

// ---------------------------------------------------------------- launch
extern "C" void kernel_launch(void* const* d_in, const int* in_sizes, int n_in,
                              void* d_out, int out_size, void* d_ws, size_t ws_size,
                              hipStream_t stream) {
    (void)in_sizes; (void)n_in; (void)out_size; (void)ws_size;

    const float* x  = (const float*)d_in[0];   // [T, H]
    const float* gw = (const float*)d_in[1];   // [H, E]
    const float* wg = (const float*)d_in[2];   // [E, H, I]
    const float* wu = (const float*)d_in[3];   // [E, H, I]
    const float* wd = (const float*)d_in[4];   // [E, I, H]
    float* out = (float*)d_out;                // [T*H] out, [1] aux

    float* w_full = (float*)d_ws;                           // T*E floats
    float* eload  = w_full + (size_t)T_TOK * E_NUM;         // 8 floats
    float* tcnt   = eload + 8;                              // 8 floats
    unsigned short* hdn = (unsigned short*)((char*)d_ws + HDN_OFF);  // T*I bf16

    moe_init_kernel<<<1024, 256, 0, stream>>>(out, eload);
    moe_router_kernel<<<T_TOK / 8, 256, 0, stream>>>(x, gw, w_full, eload, tcnt);
    moe_aux_kernel<<<1, 32, 0, stream>>>(eload, tcnt, out + (size_t)T_TOK * H_DIM);

    const dim3 g1(I_DIM / BN, T_TOK / BM);
    const dim3 g2(H_DIM / BN, T_TOK / BM);
    for (int e = 0; e < E_NUM; ++e) {
        moe_gemm1_kernel<<<g1, 256, 0, stream>>>(x, wg, wu, hdn, e);
        moe_gemm2_kernel<<<g2, 256, 0, stream>>>(hdn, wd, w_full, out, e);
    }
}
